// NewsEncoder_3393024164348
// MI455X (gfx1250) — compile-verified, matrix operand order checked by reading, not run on hardware
//
#include <hip/hip_runtime.h>
#include <hip/hip_bf16.h>
#include <math.h>

// ---------------------------------------------------------------------------
// NewsEncoder fused forward for MI455X (gfx1250, wave32, WMMA bf16)
//   x = emb[title]                       [B,S,D]
//   q,k,v = x@W{q,k,v}.T + b             [B,S,D]   -> bf16 WMMA
//   per-head softmax attention (DK=20)   -> VALU + wave32 shuffles
//   temp = tanh(ctx@Wa.T + ba)           [B,S,QD]  -> bf16 WMMA
//   w = softmax(temp@qv), out = w^T ctx  [B,D]
// ---------------------------------------------------------------------------

#define B_   8192
#define S_   32
#define D_   300
#define H_   15
#define DK_  20
#define QD_  200

#define KPAD     320          // K padded to 10 * 32
#define KSTEPS   10
#define NT_D     19           // ceil(300/16) N tiles for D-sized outputs
#define NPAD_D   304          // 19*16
#define NT_Q     13           // ceil(200/16) N tiles for QD-sized outputs
#define NPAD_Q   208          // 13*16

#define TILE_ELEMS 512                       // 32 lanes * 16 bf16 per WMMA B tile
#define PACKW_SZ   (NT_D * KSTEPS * TILE_ELEMS)   // 97280 bf16 per D-weight
#define PACKA_SZ   (NT_Q * KSTEPS * TILE_ELEMS)   // 66560 bf16 for Wa
#define PACK_TOTAL (3 * PACKW_SZ + PACKA_SZ)      // 358400 bf16

typedef __attribute__((ext_vector_type(16))) __bf16 v16bf;
typedef __attribute__((ext_vector_type(8)))  __bf16 v8bf;
typedef __attribute__((ext_vector_type(8)))  float  v8f;

// 16-bit A-fragment K mapping (ISA 7.12.2): lane<16 rows M=0..15 hold
// K = {kbase+0..7, kbase+16..23}; lanes 16..31 hold K = {+8..15, +24..31}.
// B fragments are packed in workspace in mirrored per-lane order so the load
// is 16 contiguous bf16 per lane.

__device__ __forceinline__ v16bf load_a_frag(const __bf16* base, int stride,
                                             int mt, int ks, int lane) {
  const int row = mt * 16 + (lane & 15);
  const int off = ks * 32 + ((lane & 16) ? 8 : 0);
  const __bf16* p = base + row * stride + off;
  v8bf lo = *(const v8bf*)(p);
  v8bf hi = *(const v8bf*)(p + 16);
  v16bf r;
#pragma unroll
  for (int e = 0; e < 8; ++e) { r[e] = lo[e]; r[e + 8] = hi[e]; }
  return r;
}

__device__ __forceinline__ v16bf load_b_frag(const __bf16* pack, int tile, int lane) {
  return *(const v16bf*)(pack + (tile * 32 + lane) * 16);
}

// ---------------------------------------------------------------------------
// Weight packing: f32 [nOut,300] -> bf16 B-fragment order [nt][ks][lane][e]
// ---------------------------------------------------------------------------
__global__ __launch_bounds__(256) void pack_weights_kernel(
    const float* __restrict__ Wq, const float* __restrict__ Wk,
    const float* __restrict__ Wv, const float* __restrict__ Wa,
    __bf16* __restrict__ ws) {
  int idx = blockIdx.x * blockDim.x + threadIdx.x;
  if (idx >= PACK_TOTAL) return;
  const float* W;
  __bf16* dst;
  int nOut, rel;
  if (idx < 3 * PACKW_SZ) {
    int seg = idx / PACKW_SZ;
    rel = idx - seg * PACKW_SZ;
    W = (seg == 0) ? Wq : (seg == 1) ? Wk : Wv;
    dst = ws + seg * PACKW_SZ;
    nOut = D_;
  } else {
    rel = idx - 3 * PACKW_SZ;
    W = Wa;
    dst = ws + 3 * PACKW_SZ;
    nOut = QD_;
  }
  const int e    = rel & 15;
  const int lane = (rel >> 4) & 31;
  const int tile = rel >> 9;          // tile = nt*KSTEPS + ks
  const int ks   = tile % KSTEPS;
  const int nt   = tile / KSTEPS;
  const int n = nt * 16 + (lane & 15);
  const int k = ks * 32 + e + ((e >= 8) ? 8 : 0) + ((lane >= 16) ? 8 : 0);
  float val = (n < nOut && k < D_) ? W[n * D_ + k] : 0.0f;
  dst[rel] = (__bf16)val;
}

// ---------------------------------------------------------------------------
// Fused per-batch encoder: one workgroup per batch element, 8 waves
// ---------------------------------------------------------------------------
__global__ __launch_bounds__(256, 1) void news_encoder_kernel(
    const int* __restrict__ title, const float* __restrict__ emb,
    const float* __restrict__ bq, const float* __restrict__ bk,
    const float* __restrict__ bv, const float* __restrict__ ba,
    const float* __restrict__ qvec,
    const __bf16* __restrict__ packQ, const __bf16* __restrict__ packK,
    const __bf16* __restrict__ packV, const __bf16* __restrict__ packA,
    float* __restrict__ out) {
  const int b    = blockIdx.x;
  const int tid  = threadIdx.x;
  const int lane = tid & 31;
  const int wave = tid >> 5;

  extern __shared__ char smem_raw[];
  __bf16* xb   = (__bf16*)smem_raw;                 // [32][320] bf16
  __bf16* qb   = xb + S_ * KPAD;                    // [32][304] bf16
  __bf16* kbuf = qb + S_ * NPAD_D;                  // [32][304] bf16
  __bf16* vbuf = kbuf + S_ * NPAD_D;                // [32][304] bf16
  float*  ctxf = (float*)(vbuf + S_ * NPAD_D);      // [32][304] f32
  __bf16* ctxb = (__bf16*)(ctxf + S_ * NPAD_D);     // [32][320] bf16
  float*  tempf = (float*)(ctxb + S_ * KPAD);       // [32][208] f32
  float*  attn  = tempf + S_ * NPAD_Q;              // [8][32][32] f32
  float*  wsm   = attn + 8 * 32 * 32;               // [32] f32

  // ---- Stage 0: embedding gather + f32->bf16 convert, K-pad to 320 --------
  for (int idx = tid; idx < S_ * KPAD; idx += 256) {
    const int s = idx / KPAD;
    const int c = idx - s * KPAD;
    float v = 0.0f;
    if (c < D_) {
      const int tok = title[b * S_ + s];
      v = emb[tok * D_ + c];
    }
    xb[idx] = (__bf16)v;
    ctxb[idx] = (__bf16)0.0f;   // pre-zero ctx bf16 (pad cols stay zero)
  }
  __syncthreads();

  // ---- Stage 1: q/k/v = x @ W.T + b via WMMA bf16 -------------------------
  // 3 projections x 2 M-tiles x 19 N-tiles = 114 tiles round-robin on waves
  for (int t = wave; t < 3 * 2 * NT_D; t += 8) {
    const int proj = t / (2 * NT_D);
    const int r    = t - proj * (2 * NT_D);
    const int mt   = r / NT_D;
    const int nt   = r - mt * NT_D;
    const __bf16* pw  = (proj == 0) ? packQ : (proj == 1) ? packK : packV;
    const float* bias = (proj == 0) ? bq : (proj == 1) ? bk : bv;
    __bf16* dst       = (proj == 0) ? qb : (proj == 1) ? kbuf : vbuf;
    __builtin_prefetch(pw + nt * KSTEPS * TILE_ELEMS, 0, 3);
    v8f acc = {};
#pragma unroll
    for (int ks = 0; ks < KSTEPS; ++ks) {
      v16bf a  = load_a_frag(xb, KPAD, mt, ks, lane);
      v16bf bm = load_b_frag(pw, nt * KSTEPS + ks, lane);
      acc = __builtin_amdgcn_wmma_f32_16x16x32_bf16(
          /*neg_a=*/false, a, /*neg_b=*/false, bm,
          /*c_mod=*/(short)0, acc, /*reuse_a=*/false, /*reuse_b=*/false);
    }
    const int n     = nt * 16 + (lane & 15);
    const int mbase = mt * 16 + ((lane & 16) ? 8 : 0);
    const float bn  = (n < D_) ? bias[n] : 0.0f;
#pragma unroll
    for (int e = 0; e < 8; ++e) {
      const float val = (n < D_) ? (acc[e] + bn) : 0.0f;
      dst[(mbase + e) * NPAD_D + n] = (__bf16)val;
    }
  }
  __syncthreads();

  // ---- Stage 2: per-head scaled-dot-product attention (DK=20, S=32) -------
  const float scale = 0.22360679775f;  // 1/sqrt(20)
  for (int h = wave; h < H_; h += 8) {
    float* abuf = attn + wave * 32 * 32;
    float kreg[DK_];
#pragma unroll
    for (int d = 0; d < DK_; ++d)
      kreg[d] = (float)kbuf[lane * NPAD_D + h * DK_ + d];  // lane = key index j
    for (int i = 0; i < S_; ++i) {
      float s = 0.0f;
#pragma unroll
      for (int d = 0; d < DK_; ++d)
        s += (float)qb[i * NPAD_D + h * DK_ + d] * kreg[d];
      s *= scale;
      float mx = s;
      for (int o = 16; o > 0; o >>= 1) mx = fmaxf(mx, __shfl_xor(mx, o, 32));
      const float ex = __expf(s - mx);
      float sm = ex;
      for (int o = 16; o > 0; o >>= 1) sm += __shfl_xor(sm, o, 32);
      abuf[i * 32 + lane] = ex / sm;
    }
    // ctx rows: lane = query index i
#pragma unroll
    for (int d = 0; d < DK_; ++d) {
      float acc = 0.0f;
      for (int j = 0; j < S_; ++j)
        acc += abuf[lane * 32 + j] * (float)vbuf[j * NPAD_D + h * DK_ + d];
      ctxf[lane * NPAD_D + h * DK_ + d] = acc;
      ctxb[lane * KPAD + h * DK_ + d]   = (__bf16)acc;
    }
  }
  __syncthreads();

  // ---- Stage 3: temp = tanh(ctx @ Wa.T + ba) via WMMA bf16 ----------------
  for (int t = wave; t < 2 * NT_Q; t += 8) {
    const int mt = t / NT_Q;
    const int nt = t - mt * NT_Q;
    v8f acc = {};
#pragma unroll
    for (int ks = 0; ks < KSTEPS; ++ks) {
      v16bf a  = load_a_frag(ctxb, KPAD, mt, ks, lane);
      v16bf bm = load_b_frag(packA, nt * KSTEPS + ks, lane);
      acc = __builtin_amdgcn_wmma_f32_16x16x32_bf16(
          false, a, false, bm, (short)0, acc, false, false);
    }
    const int n     = nt * 16 + (lane & 15);
    const int mbase = mt * 16 + ((lane & 16) ? 8 : 0);
    const float bn  = (n < QD_) ? ba[n] : 0.0f;
#pragma unroll
    for (int e = 0; e < 8; ++e) {
      const float val = (n < QD_) ? tanhf(acc[e] + bn) : 0.0f;
      tempf[(mbase + e) * NPAD_Q + n] = val;
    }
  }
  __syncthreads();

  // ---- Stage 4: w = softmax(temp @ qv) over S (wave 0, lane = s) ----------
  if (wave == 0) {
    float s = 0.0f;
    for (int qd = 0; qd < QD_; ++qd)
      s += tempf[lane * NPAD_Q + qd] * qvec[qd];
    float mx = s;
    for (int o = 16; o > 0; o >>= 1) mx = fmaxf(mx, __shfl_xor(mx, o, 32));
    const float ex = __expf(s - mx);
    float sm = ex;
    for (int o = 16; o > 0; o >>= 1) sm += __shfl_xor(sm, o, 32);
    wsm[lane] = ex / sm;
  }
  __syncthreads();

  // ---- Stage 5: out[b,:] = sum_s w[s] * ctx[s,:] --------------------------
  for (int d = tid; d < D_; d += 256) {
    float o = 0.0f;
#pragma unroll
    for (int s = 0; s < S_; ++s) o += wsm[s] * ctxf[s * NPAD_D + d];
    out[b * D_ + d] = o;
  }
}

// ---------------------------------------------------------------------------
extern "C" void kernel_launch(void* const* d_in, const int* in_sizes, int n_in,
                              void* d_out, int out_size, void* d_ws, size_t ws_size,
                              hipStream_t stream) {
  (void)in_sizes; (void)n_in; (void)out_size; (void)ws_size;
  const int*   title = (const int*)d_in[0];
  const float* emb   = (const float*)d_in[1];
  const float* Wq    = (const float*)d_in[2];
  const float* bq    = (const float*)d_in[3];
  const float* Wk    = (const float*)d_in[4];
  const float* bk    = (const float*)d_in[5];
  const float* Wv    = (const float*)d_in[6];
  const float* bv    = (const float*)d_in[7];
  const float* Wa    = (const float*)d_in[8];
  const float* ba    = (const float*)d_in[9];
  const float* qvec  = (const float*)d_in[10];
  float* out = (float*)d_out;

  __bf16* ws    = (__bf16*)d_ws;
  __bf16* packQ = ws;
  __bf16* packK = ws + PACKW_SZ;
  __bf16* packV = ws + 2 * PACKW_SZ;
  __bf16* packA = ws + 3 * PACKW_SZ;

  pack_weights_kernel<<<(PACK_TOTAL + 255) / 256, 256, 0, stream>>>(
      Wq, Wk, Wv, Wa, ws);

  // Dynamic LDS: xb + q/k/v + ctxf + ctxb + tempf + attn + wsm
  const size_t lds =
      (size_t)S_ * KPAD * 2 +           // xb
      3 * (size_t)S_ * NPAD_D * 2 +     // q, k, v (bf16)
      (size_t)S_ * NPAD_D * 4 +         // ctx f32
      (size_t)S_ * KPAD * 2 +           // ctx bf16 (K-padded)
      (size_t)S_ * NPAD_Q * 4 +         // temp f32
      (size_t)8 * 32 * 32 * 4 +         // per-wave attn scratch
      32 * 4;                           // softmax weights
  (void)hipFuncSetAttribute((const void*)news_encoder_kernel,
                            hipFuncAttributeMaxDynamicSharedMemorySize,
                            (int)lds);

  news_encoder_kernel<<<B_, 256, lds, stream>>>(
      title, emb, bq, bk, bv, ba, qvec, packQ, packK, packV, packA, out);
}
